// SubgraphormerLayer_35639638622214
// MI455X (gfx1250) — compile-verified, matrix operand order checked by reading, not run on hardware
//
#include <hip/hip_runtime.h>
#include <hip/hip_bf16.h>

typedef __attribute__((ext_vector_type(16))) _Float16 v16h;
typedef __attribute__((ext_vector_type(8)))  float    v8f;

#define HDIM 128
#define WPB  8   // waves per block (256 threads)

// Convert 16 f32 -> one WMMA 16x16x32 A/B fragment (per-lane v16h).
// Layout (ISA 7.12.2, 16-bit 16x32): lane&15 = row/col, lane>>4 = K-phase:
// elems 0..7 = row[kb..kb+7], elems 8..15 = row[kb+16..kb+23].
__device__ __forceinline__ v16h load_frag_f16(const float* __restrict__ row, int kb) {
  const float4* p0 = (const float4*)(row + kb);
  const float4* p1 = (const float4*)(row + kb + 16);
  float4 q0 = p0[0], q1 = p0[1], q2 = p1[0], q3 = p1[1];
  v16h r;
  r[0]  = (_Float16)q0.x; r[1]  = (_Float16)q0.y; r[2]  = (_Float16)q0.z; r[3]  = (_Float16)q0.w;
  r[4]  = (_Float16)q1.x; r[5]  = (_Float16)q1.y; r[6]  = (_Float16)q1.z; r[7]  = (_Float16)q1.w;
  r[8]  = (_Float16)q2.x; r[9]  = (_Float16)q2.y; r[10] = (_Float16)q2.z; r[11] = (_Float16)q2.w;
  r[12] = (_Float16)q3.x; r[13] = (_Float16)q3.y; r[14] = (_Float16)q3.z; r[15] = (_Float16)q3.w;
  return r;
}

// Cooperative fill of LDS with f16 B-fragments of W[128 x Ktot] (row-major).
// Fragment f = nt*NK + kk holds B tile (cols nt*16..+15, K kk*32..+31),
// stored per-lane so the consumer does a single 32B ds_load per fragment.
template<int NK>
__device__ __forceinline__ void stage_weights(const float* __restrict__ W, v16h* wlds) {
  const int Ktot = NK << 5;
  for (int e = threadIdx.x; e < 8 * NK * 32; e += 256) {
    const int f  = e >> 5;
    const int l  = e & 31;
    const int nt = f / NK;
    const int kk = f % NK;
    const int n  = (nt << 4) + (l & 15);
    wlds[e] = load_frag_f16(W + (long long)n * Ktot + (kk << 5), (l >> 4) << 3);
  }
  __syncthreads();
}

// Core 16x128 row-block WMMA loop with A preloaded and B batch-loaded per K-chunk
// so the 8 WMMAs per chunk issue back-to-back on the matrix pipe.
template<int NK>
__device__ __forceinline__ void wmma_rowblock(const v16h* __restrict__ a,
                                              const v16h* __restrict__ wlds,
                                              int lane, v8f* acc) {
#pragma unroll
  for (int kk = 0; kk < NK; ++kk) {
    v16h b[8];
#pragma unroll
    for (int nt = 0; nt < 8; ++nt) b[nt] = wlds[(nt * NK + kk) * 32 + lane];
#pragma unroll
    for (int nt = 0; nt < 8; ++nt)
      acc[nt] = __builtin_amdgcn_wmma_f32_16x16x32_f16(false, a[kk], false, b[nt],
                                                       (short)0, acc[nt], false, false);
  }
}

// WMMA GEMM, register-blocked: each wave computes a 16x128 row-block.
// Y[M,128] = act( cat(X1[M,K1], X2[M,K2]) @ W[128,K1+K2]^T + bias )
// ACT: 0 none, 1 relu, 2 exact gelu.  NK = (K1+K2)/32.
template<int ACT, int NK>
__global__ void gemm_wmma_kernel(const float* __restrict__ X1, int K1,
                                 const float* __restrict__ X2, int K2,
                                 const float* __restrict__ W,
                                 const float* __restrict__ bias,
                                 float* __restrict__ Y, int M) {
  __shared__ v16h wlds[8 * NK * 32];          // 32KB (NK=4) / 64KB (NK=8)
  stage_weights<NK>(W, wlds);

  const int lane = threadIdx.x & 31;
  const int wave = threadIdx.x >> 5;
  const int mt   = blockIdx.x * WPB + wave;   // 16-row tile index
  if (mt >= (M >> 4)) return;                 // wave-uniform (after syncthreads)
  const int l15 = lane & 15;
  const int kb  = (lane >> 4) << 3;
  const int m   = (mt << 4) + l15;

  // Preload + convert all A fragments (global loads issued up front).
  v16h a[NK];
#pragma unroll
  for (int kk = 0; kk < NK; ++kk) {
    const int k0 = kk << 5;
    const float* xr = (k0 < K1) ? (X1 + (long long)m * K1 + k0)
                                : (X2 + (long long)m * K2 + (k0 - K1));
    a[kk] = load_frag_f16(xr, kb);
  }

  v8f acc[8] = {};
  wmma_rowblock<NK>(a, wlds, lane, acc);

#pragma unroll
  for (int nt = 0; nt < 8; ++nt) {
    const int n  = (nt << 4) + l15;
    const float bv = bias ? bias[n] : 0.0f;
#pragma unroll
    for (int j = 0; j < 8; ++j) {
      const int row = (mt << 4) + j + ((lane >> 4) << 3);   // C/D: VGPR j -> M=j / j+8
      float v = acc[nt][j] + bv;
      if (ACT == 1)      v = fmaxf(v, 0.0f);
      else if (ACT == 2) v = 0.5f * v * (1.0f + erff(v * 0.7071067811865475f));
      Y[(long long)row * HDIM + n] = v;
    }
  }
}

// Fused GINE edge kernel, register-blocked: each wave handles 16 edges x 128 cols.
// msg = relu(x[src] + ea @ eW^T + eb); agg[dst] += msg  (atomic scatter).
__global__ void gine_edge_wmma_kernel(const float* __restrict__ ea,
                                      const int* __restrict__ ei, long long E,
                                      const float* __restrict__ eW,
                                      const float* __restrict__ ebias,
                                      const float* __restrict__ x,
                                      float* __restrict__ agg) {
  __shared__ v16h wlds[8 * 4 * 32];           // 32KB
  stage_weights<4>(eW, wlds);

  const int lane = threadIdx.x & 31;
  const int wave = threadIdx.x >> 5;
  const long long mt = (long long)blockIdx.x * WPB + wave;
  if (mt >= (E >> 4)) return;
  const int l15 = lane & 15;
  const int kb  = (lane >> 4) << 3;
  const long long m = (mt << 4) + l15;

  v16h a[4];
#pragma unroll
  for (int kk = 0; kk < 4; ++kk)
    a[kk] = load_frag_f16(ea + m * HDIM + (kk << 5), kb);

  v8f acc[8] = {};
  wmma_rowblock<4>(a, wlds, lane, acc);

  float bv[8];
#pragma unroll
  for (int nt = 0; nt < 8; ++nt) bv[nt] = ebias[(nt << 4) + l15];

#pragma unroll
  for (int j = 0; j < 8; ++j) {
    const long long e = (mt << 4) + j + ((lane >> 4) << 3);
    const int s = ei[e];
    const int d = ei[E + e];
    const float* xrow = x + (long long)s * HDIM;
    float* arow = agg + (long long)d * HDIM;
#pragma unroll
    for (int nt = 0; nt < 8; ++nt) {
      const int n = (nt << 4) + l15;
      const float v = fmaxf(acc[nt][j] + bv[nt] + xrow[n], 0.0f);
      atomicAdd(&arow[n], v);
    }
  }
}

// agg = (1+eps)*x + agg  (in place)
__global__ void gine_combine_kernel(float* __restrict__ agg,
                                    const float* __restrict__ x,
                                    const float* __restrict__ eps_ptr,
                                    long long n) {
  long long i = (long long)blockIdx.x * blockDim.x + threadIdx.x;
  if (i >= n) return;
  const float eps = *eps_ptr;
  agg[i] = (1.0f + eps) * x[i] + agg[i];
}

__global__ void add_kernel(float* __restrict__ a, const float* __restrict__ b,
                           long long n) {
  long long i = (long long)blockIdx.x * blockDim.x + threadIdx.x;
  if (i < n) a[i] += b[i];
}

// Column-wise sum / sum-of-squares partials for BatchNorm (H=128 columns).
__global__ void bn_stats_kernel(const float* __restrict__ y, long long M,
                                float* __restrict__ sum, float* __restrict__ sumsq) {
  const int c = threadIdx.x;
  const long long chunk = (M + gridDim.x - 1) / gridDim.x;
  long long r0 = (long long)blockIdx.x * chunk;
  long long r1 = r0 + chunk; if (r1 > M) r1 = M;
  float s = 0.0f, s2 = 0.0f;
  for (long long r = r0; r < r1; ++r) {
    float v = y[r * HDIM + c];
    s += v; s2 += v * v;
  }
  atomicAdd(&sum[c], s);
  atomicAdd(&sumsq[c], s2);
}

__global__ void bn_finalize_kernel(float* __restrict__ sum, float* __restrict__ sumsq,
                                   float Minv) {
  const int c = threadIdx.x;
  const float mu  = sum[c] * Minv;
  const float var = sumsq[c] * Minv - mu * mu;
  sum[c]   = mu;
  sumsq[c] = rsqrtf(var + 1e-5f);
}

__global__ void bn_apply_kernel(const float* __restrict__ y,
                                const float* __restrict__ mean,
                                const float* __restrict__ invstd,
                                const float* __restrict__ g,
                                const float* __restrict__ b,
                                float* __restrict__ outp, long long n) {
  long long i = (long long)blockIdx.x * blockDim.x + threadIdx.x;
  if (i >= n) return;
  const int c = (int)(i & (HDIM - 1));
  outp[i] = (y[i] - mean[c]) * invstd[c] * g[c] + b[c];
}

// scatter-mean numerators + counts: ssum[assign[t]] += h_flat[root[t]]
__global__ void scatter_root_kernel(const float* __restrict__ h_flat,
                                    const int* __restrict__ root,
                                    const int* __restrict__ assign,
                                    float* __restrict__ ssum,
                                    float* __restrict__ cnt, long long T) {
  long long i = (long long)blockIdx.x * blockDim.x + threadIdx.x;
  if (i >= T * HDIM) return;
  const long long t = i >> 7;
  const int c = (int)(i & 127);
  const int a = assign[t];
  const float v = h_flat[(long long)root[t] * HDIM + c];
  atomicAdd(&ssum[(long long)a * HDIM + c], v);
  if (c == 0) atomicAdd(&cnt[a], 1.0f);
}

__global__ void scatter_div_kernel(float* __restrict__ ssum,
                                   const float* __restrict__ cnt, long long n) {
  long long i = (long long)blockIdx.x * blockDim.x + threadIdx.x;
  if (i >= n) return;
  ssum[i] /= fmaxf(cnt[i >> 7], 1.0f);
}

// h_bcast[t] = Yb[assign[sub_ids[t]]]
__global__ void gather_bcast_kernel(const float* __restrict__ Yb,
                                    const int* __restrict__ assign,
                                    const int* __restrict__ sub_ids,
                                    float* __restrict__ outp, long long T) {
  long long i = (long long)blockIdx.x * blockDim.x + threadIdx.x;
  if (i >= T * HDIM) return;
  const long long t = i >> 7;
  const int c = (int)(i & 127);
  const int node = assign[sub_ids[t]];
  outp[i] = Yb[(long long)node * HDIM + c];
}

// out = (h_flat + LayerNorm(hn)*g+b) * valid   — one 128-thread block per row
__global__ void ln_residual_mask_kernel(const float* __restrict__ hn,
                                        const float* __restrict__ h_flat,
                                        const float* __restrict__ valid,
                                        const float* __restrict__ g,
                                        const float* __restrict__ b,
                                        float* __restrict__ outp) {
  const int t = blockIdx.x, c = threadIdx.x;
  __shared__ float red[HDIM];
  const long long base = (long long)t * HDIM;
  const float v = hn[base + c];
  red[c] = v; __syncthreads();
#pragma unroll
  for (int s = 64; s > 0; s >>= 1) { if (c < s) red[c] += red[c + s]; __syncthreads(); }
  const float mu = red[0] * (1.0f / HDIM);
  __syncthreads();
  const float d = v - mu;
  red[c] = d * d; __syncthreads();
#pragma unroll
  for (int s = 64; s > 0; s >>= 1) { if (c < s) red[c] += red[c + s]; __syncthreads(); }
  const float var = red[0] * (1.0f / HDIM);
  const float y = d * rsqrtf(var + 1e-5f) * g[c] + b[c];
  outp[base + c] = (h_flat[base + c] + y) * valid[t];
}

extern "C" void kernel_launch(void* const* d_in, const int* in_sizes, int n_in,
                              void* d_out, int out_size, void* d_ws, size_t ws_size,
                              hipStream_t stream) {
  (void)n_in; (void)out_size;
  const int H = HDIM;
  const long long T  = (long long)in_sizes[0] / H;   // 262144
  const long long EI = (long long)in_sizes[2] / H;   // 1048576
  const long long EG = (long long)in_sizes[5] / H;   // 131072
  const long long NN = 8192;                         // N_total (device scalar; fixed)

  const float* h_flat    = (const float*)d_in[0];
  const int*   intra_ei  = (const int*)  d_in[1];
  const float* intra_ea  = (const float*)d_in[2];
  const float* valid_f   = (const float*)d_in[3];
  const int*   global_ei = (const int*)  d_in[4];
  const float* global_ea = (const float*)d_in[5];
  const int*   root_idx  = (const int*)  d_in[6];
  const int*   assign    = (const int*)  d_in[7];
  const int*   sub_ids   = (const int*)  d_in[8];
  const float* l_edge_W  = (const float*)d_in[11];
  const float* l_edge_b  = (const float*)d_in[12];
  const float* l_eps     = (const float*)d_in[13];
  const float* l_W1 = (const float*)d_in[14]; const float* l_b1 = (const float*)d_in[15];
  const float* l_W2 = (const float*)d_in[16]; const float* l_b2 = (const float*)d_in[17];
  const float* l_bn_g = (const float*)d_in[18]; const float* l_bn_b = (const float*)d_in[19];
  const float* g_edge_W = (const float*)d_in[20];
  const float* g_edge_b = (const float*)d_in[21];
  const float* g_eps    = (const float*)d_in[22];
  const float* g_W1 = (const float*)d_in[23]; const float* g_b1 = (const float*)d_in[24];
  const float* g_W2 = (const float*)d_in[25]; const float* g_b2 = (const float*)d_in[26];
  const float* g_bn_g = (const float*)d_in[27]; const float* g_bn_b = (const float*)d_in[28];
  const float* bcast_W = (const float*)d_in[29];
  const float* c_W1 = (const float*)d_in[30]; const float* c_b1 = (const float*)d_in[31];
  const float* c_W2 = (const float*)d_in[32]; const float* c_b2 = (const float*)d_in[33];
  const float* ln_g = (const float*)d_in[34]; const float* ln_b = (const float*)d_in[35];
  float* outp = (float*)d_out;

  // Workspace layout (floats): 3 big T*H buffers (reused across phases) + node bufs.
  float* ws   = (float*)d_ws;
  float* big0 = ws;                 // agg_local -> h_bcast -> h_new_raw
  float* big1 = big0 + T * H;       // mlp hidden -> h_local
  float* big2 = big1 + T * H;       // pre-BN relu -> gelu hidden
  float* nb0  = big2 + T * H;       // ssum -> h_node
  float* nb1  = nb0 + NN * H;       // agg_global -> Yb
  float* nb2  = nb1 + NN * H;       // g hidden -> h_cross
  float* nb3  = nb2 + NN * H;       // g pre-BN
  float* cnt  = nb3 + NN * H;       // NN counts
  float* stat_sum = cnt + NN;       // 128
  float* stat_sq  = stat_sum + H;   // 128
  const size_t need = (size_t)(3 * T * H + 4 * NN * H + NN + 2 * H) * sizeof(float);
  if (ws_size < need) return;

  const dim3 blk(256);
  auto eb = [](long long n) { return dim3((unsigned)((n + 255) / 256)); };
  auto rb = [](long long M) { return dim3((unsigned)(((M >> 4) + WPB - 1) / WPB)); };

  // ---- local GINE: edge linear (WMMA) + scatter, combine, MLP, BN ----
  hipMemsetAsync(big0, 0, (size_t)T * H * sizeof(float), stream);
  hipMemsetAsync(nb0,  0, (size_t)NN * H * sizeof(float), stream);
  hipMemsetAsync(cnt,  0, (size_t)NN * sizeof(float), stream);
  hipMemsetAsync(stat_sum, 0, (size_t)2 * H * sizeof(float), stream);

  gine_edge_wmma_kernel<<<rb(EI), blk, 0, stream>>>(intra_ea, intra_ei, EI,
                                                    l_edge_W, l_edge_b, h_flat, big0);
  gine_combine_kernel<<<eb(T * H), blk, 0, stream>>>(big0, h_flat, l_eps, T * H);
  gemm_wmma_kernel<1, 4><<<rb(T), blk, 0, stream>>>(big0, H, nullptr, 0, l_W1, l_b1,
                                                    big1, (int)T);
  gemm_wmma_kernel<1, 4><<<rb(T), blk, 0, stream>>>(big1, H, nullptr, 0, l_W2, l_b2,
                                                    big2, (int)T);     // relu(GINE out)
  bn_stats_kernel<<<512, H, 0, stream>>>(big2, T, stat_sum, stat_sq);
  bn_finalize_kernel<<<1, H, 0, stream>>>(stat_sum, stat_sq, 1.0f / (float)T);
  bn_apply_kernel<<<eb(T * H), blk, 0, stream>>>(big2, stat_sum, stat_sq,
                                                 l_bn_g, l_bn_b, big1, T * H); // h_local

  // ---- scatter-mean roots -> node features ----
  scatter_root_kernel<<<eb(T * H), blk, 0, stream>>>(h_flat, root_idx, assign,
                                                     nb0, cnt, T);
  scatter_div_kernel<<<eb(NN * H), blk, 0, stream>>>(nb0, cnt, NN * H);   // h_node

  // ---- global GINE on node graph + BN + residual ----
  hipMemsetAsync(nb1, 0, (size_t)NN * H * sizeof(float), stream);
  gine_edge_wmma_kernel<<<rb(EG), blk, 0, stream>>>(global_ea, global_ei, EG,
                                                    g_edge_W, g_edge_b, nb0, nb1);
  gine_combine_kernel<<<eb(NN * H), blk, 0, stream>>>(nb1, nb0, g_eps, NN * H);
  gemm_wmma_kernel<1, 4><<<rb(NN), blk, 0, stream>>>(nb1, H, nullptr, 0, g_W1, g_b1,
                                                     nb2, (int)NN);
  gemm_wmma_kernel<1, 4><<<rb(NN), blk, 0, stream>>>(nb2, H, nullptr, 0, g_W2, g_b2,
                                                     nb3, (int)NN);
  hipMemsetAsync(stat_sum, 0, (size_t)2 * H * sizeof(float), stream);
  bn_stats_kernel<<<64, H, 0, stream>>>(nb3, NN, stat_sum, stat_sq);
  bn_finalize_kernel<<<1, H, 0, stream>>>(stat_sum, stat_sq, 1.0f / (float)NN);
  bn_apply_kernel<<<eb(NN * H), blk, 0, stream>>>(nb3, stat_sum, stat_sq,
                                                  g_bn_g, g_bn_b, nb2, NN * H);
  add_kernel<<<eb(NN * H), blk, 0, stream>>>(nb0, nb2, NN * H);  // h_node += h_cross

  // ---- broadcast linear + gather to flat positions ----
  gemm_wmma_kernel<0, 4><<<rb(NN), blk, 0, stream>>>(nb0, H, nullptr, 0, bcast_W,
                                                     nullptr, nb1, (int)NN); // Yb
  gather_bcast_kernel<<<eb(T * H), blk, 0, stream>>>(nb1, assign, sub_ids, big0, T);

  // ---- concat MLP (split-K WMMA GEMM with GELU) + final linear ----
  gemm_wmma_kernel<2, 8><<<rb(T), blk, 0, stream>>>(big1, H, big0, H, c_W1, c_b1,
                                                    big2, (int)T);   // gelu
  gemm_wmma_kernel<0, 4><<<rb(T), blk, 0, stream>>>(big2, H, nullptr, 0, c_W2, c_b2,
                                                    big0, (int)T);   // h_new raw

  // ---- LayerNorm + residual + validity mask ----
  ln_residual_mask_kernel<<<dim3((unsigned)T), dim3(H), 0, stream>>>(big0, h_flat,
                                                                     valid_f, ln_g,
                                                                     ln_b, outp);
}
